// WindowAttention2D_19181323944193
// MI455X (gfx1250) — matrix operand order repaired and checked by reading, not compiled, not run on hardware
//
#include <hip/hip_runtime.h>
#include <hip/hip_bf16.h>

typedef __bf16 bf16;
typedef __attribute__((ext_vector_type(16))) __bf16 v16bf;
typedef __attribute__((ext_vector_type(8)))  float  v8f;

union AFrag { v16bf v; unsigned u[8]; };

// Generic pointers to LDS carry the LDS byte offset in bits [31:0]
// (ISA 10.2), so truncation yields the DS address.
__device__ __forceinline__ unsigned lds_u32(const void* p) {
    return (unsigned)(uintptr_t)p;
}

// CDNA5 async global->LDS copy (ASYNCcnt-tracked), GVS mode:
// mem_addr = SADDR(base) + VADDR(byte offset) ; LDS addr from VDST vgpr.
__device__ __forceinline__ void async_b128(unsigned lds, unsigned goff,
                                           const void* base) {
    asm volatile("global_load_async_to_lds_b128 %0, %1, %2"
                 :: "v"(lds), "v"(goff), "s"(base) : "memory");
}
__device__ __forceinline__ void async_b32(unsigned lds, unsigned goff,
                                          const void* base) {
    asm volatile("global_load_async_to_lds_b32 %0, %1, %2"
                 :: "v"(lds), "v"(goff), "s"(base) : "memory");
}
__device__ __forceinline__ void wait_async0() {
    asm volatile("s_wait_asynccnt 0" ::: "memory");
}

// ---------------------------------------------------------------------------
// fp32 -> bf16 conversion
// ---------------------------------------------------------------------------
__global__ void f32_to_bf16_kernel(const float* __restrict__ in,
                                   bf16* __restrict__ out, int n) {
    int i = blockIdx.x * 256 + threadIdx.x;
    if (i < n) out[i] = (bf16)in[i];
}

// ---------------------------------------------------------------------------
// Tiled bf16 WMMA GEMM:  C[m,n] = sum_k A[m,k] * B[k,n] + bias[m]
// A: (M,K) bf16 weights, B: (batch,K,N) bf16, C: (batch,M,N)
// Block tile 64x128, K-chunk 32, 256 threads = 8 waves, 4 16x16 tiles/wave.
// Double-buffered async global->LDS staging.
// ---------------------------------------------------------------------------
template <bool OUTF32>
__global__ __launch_bounds__(256)
void gemm_wmma_bf16(const bf16* __restrict__ A, const bf16* __restrict__ B,
                    const float* __restrict__ bias, bf16* __restrict__ Cb,
                    float* __restrict__ Cf, int M, int K, int N) {
    __shared__ bf16 Wt[2][64][40];   // 80B row stride (16B aligned)
    __shared__ bf16 Xt[2][32][136];  // 272B row stride (16B aligned)

    const int tid  = threadIdx.x;
    const int lane = tid & 31;
    const int wv   = tid >> 5;
    const int n0   = blockIdx.x * 128;
    const int m0   = blockIdx.y * 64;
    const bf16* Bb = B + (size_t)blockIdx.z * (size_t)K * (size_t)N;
    const size_t batchC = (size_t)blockIdx.z * (size_t)M * (size_t)N;

    const int mi  = wv & 3;          // M tile (0..3)
    const int njb = (wv >> 2) * 4;   // first of 4 N tiles (0 or 4)

    const int col   = lane & 15;
    const int hic   = lane >> 4;
    const int kbA   = hic << 4;
    const int kbB   = hic << 4;
    const int rbase = hic << 3;

    // staging coordinates (per thread)
    const int wrow = tid >> 2, wko = (tid & 3) * 8;   // 64x32 weight tile
    const int xrow = tid >> 4, xno = (tid & 15) * 8;  // 32x128 act tile (x2)

    auto stage = [&](int buf, int kc) {
        async_b128(lds_u32(&Wt[buf][wrow][wko]),
                   (unsigned)(((size_t)(m0 + wrow) * K + kc + wko) * 2), A);
        async_b128(lds_u32(&Xt[buf][xrow][xno]),
                   (unsigned)(((size_t)(kc + xrow) * N + n0 + xno) * 2), Bb);
        async_b128(lds_u32(&Xt[buf][xrow + 16][xno]),
                   (unsigned)(((size_t)(kc + xrow + 16) * N + n0 + xno) * 2), Bb);
    };

    v8f acc[4];
#pragma unroll
    for (int j = 0; j < 4; ++j) acc[j] = (v8f){0.f,0.f,0.f,0.f,0.f,0.f,0.f,0.f};

    const int NK = K / 32;
    stage(0, 0);
    for (int ik = 0; ik < NK; ++ik) {
        wait_async0();       // my async copies for chunk ik landed
        __syncthreads();     // everyone's landed; prev compute done
        const int buf = ik & 1;
        if (ik + 1 < NK) stage(buf ^ 1, (ik + 1) * 32);  // overlap next copy

        AFrag a;  // A 16x32 fragment (contiguous k = 16*hic + s, same as B)
        {
            int row = mi * 16 + (lane & 15);
#pragma unroll
            for (int v = 0; v < 8; ++v) {
                int k0 = kbA + (v << 1);
                a.u[v] = *(const unsigned*)&Wt[buf][row][k0];
            }
        }
#pragma unroll
        for (int j = 0; j < 4; ++j) {
            v16bf bfr;
#pragma unroll
            for (int v = 0; v < 8; ++v) {
                int kk = kbB + 2 * v;
                bfr[2 * v]     = Xt[buf][kk][(njb + j) * 16 + col];
                bfr[2 * v + 1] = Xt[buf][kk + 1][(njb + j) * 16 + col];
            }
            acc[j] = __builtin_amdgcn_wmma_f32_16x16x32_bf16(
                         false, a.v, false, bfr, (short)0, acc[j], false, false);
        }
    }

#pragma unroll
    for (int v = 0; v < 8; ++v) {
        int row = m0 + mi * 16 + rbase + v;
        float bb = bias[row];
        size_t cbase = batchC + (size_t)row * N + n0;
#pragma unroll
        for (int j = 0; j < 4; ++j) {
            size_t idx = cbase + (njb + j) * 16 + col;
            if (OUTF32) Cf[idx] = acc[j][v] + bb;
            else        Cb[idx] = (bf16)(acc[j][v] + bb);
        }
    }
}

// ---------------------------------------------------------------------------
// Window attention: one block per (batch, window, head). 256 threads.
// qkv: (b, 3*192, 192, 192) bf16 ; out: (b, 192, 192, 192) bf16
// ---------------------------------------------------------------------------
#define QSTR 264  // LDS row stride (bf16 elems) for channel-major Q/K/V & P

__global__ __launch_bounds__(256)
void attn_wmma_kernel(const bf16* __restrict__ qkv,
                      const float* __restrict__ rpe,
                      bf16* __restrict__ outb) {
    extern __shared__ char smem[];
    bf16*  Qs   = (bf16*)smem;            // [32][QSTR] channel-major
    bf16*  Ks   = Qs + 32 * QSTR;         // [32][QSTR]
    bf16*  Vs   = Ks + 32 * QSTR;         // [32][QSTR]
    bf16*  Ps   = Vs + 32 * QSTR;         // [8 waves][16][QSTR] bf16 probs
    float* rpeS = (float*)(Ps + 8 * 16 * QSTR);  // [961]

    const int tid  = threadIdx.x;
    const int lane = tid & 31;
    const int wv   = tid >> 5;

    int id = blockIdx.x;
    const int head = id % 6;  id /= 6;
    const int wd   = id % 12; id /= 12;
    const int hd   = id % 12; id /= 12;
    const int bi   = id;

    // async-stage Q,K,V (32 channels x 256 positions each, channel-major)
    {
        const size_t cs = 36864;  // 192*192
        for (int it = 0; it < 12; ++it) {
            int l = it * 256 + tid;
            int half = l & 1, wi = (l >> 1) & 15, c = (l >> 5) & 31, mat = l >> 10;
            size_t g = ((size_t)bi * 576 + (size_t)mat * 192 + head * 32 + c) * cs
                     + (size_t)(hd * 16 + wi) * 192 + wd * 16 + half * 8;
            bf16* dst = (mat == 0) ? Qs : ((mat == 1) ? Ks : Vs);
            async_b128(lds_u32(dst + c * QSTR + wi * 16 + half * 8),
                       (unsigned)(g * 2), qkv);
        }
    }
    // async-stage this head's RPE table
    for (int i = tid; i < 961; i += 256)
        async_b32(lds_u32(rpeS + i), (unsigned)((head * 961 + i) * 4), rpe);

    wait_async0();
    __syncthreads();

    const float scale = 0.17677669529663687f;  // 1/sqrt(32)
    const int col   = lane & 15;
    const int hic   = lane >> 4;
    const int kbA   = hic << 4;
    const int kbB   = hic << 4;
    const int rbase = hic << 3;

    for (int sp = wv; sp < 16; sp += 8) {  // two 16-row strips per wave
        const int q0 = sp * 16;

        // A fragment of Q (16 rows x K=32 channels)
        AFrag aq;
        {
            int row = q0 + (lane & 15);
#pragma unroll
            for (int v = 0; v < 8; ++v) {
                int k0 = kbA + (v << 1);
                aq.v[2 * v]     = Qs[k0 * QSTR + row];
                aq.v[2 * v + 1] = Qs[(k0 + 1) * QSTR + row];
            }
        }

        // S = Q @ K^T : 16 column tiles, full 16x256 strip in registers
        v8f acc[16];
#pragma unroll
        for (int jt = 0; jt < 16; ++jt) {
            v16bf bk;
#pragma unroll
            for (int v = 0; v < 8; ++v) {
                int kk = kbB + 2 * v;
                bk[2 * v]     = Ks[kk * QSTR + jt * 16 + col];
                bk[2 * v + 1] = Ks[(kk + 1) * QSTR + jt * 16 + col];
            }
            v8f z = {0.f,0.f,0.f,0.f,0.f,0.f,0.f,0.f};
            acc[jt] = __builtin_amdgcn_wmma_f32_16x16x32_bf16(
                          false, aq.v, false, bk, (short)0, z, false, false);
        }

        // scale + RPE bias + row max, all in accumulator layout:
        // row = q0+rbase+v, col = jt*16+col ; bias = (jt-sp+15)*31 + (col-row%16+15)
        float rmax[8];
#pragma unroll
        for (int v = 0; v < 8; ++v) rmax[v] = -1e30f;
#pragma unroll
        for (int jt = 0; jt < 16; ++jt) {
            const int bh = (jt - sp + 15) * 31;
#pragma unroll
            for (int v = 0; v < 8; ++v) {
                float s = acc[jt][v] * scale + rpeS[bh + (col - rbase - v + 15)];
                acc[jt][v] = s;
                rmax[v] = fmaxf(rmax[v], s);
            }
        }
#pragma unroll
        for (int m = 1; m < 16; m <<= 1) {
#pragma unroll
            for (int v = 0; v < 8; ++v)
                rmax[v] = fmaxf(rmax[v], __shfl_xor(rmax[v], m, 32));
        }

        // exp, row sums, write P (bf16) to per-wave LDS strip
        float rsum[8];
#pragma unroll
        for (int v = 0; v < 8; ++v) rsum[v] = 0.f;
        bf16* Pw = Ps + wv * 16 * QSTR;
#pragma unroll
        for (int jt = 0; jt < 16; ++jt) {
#pragma unroll
            for (int v = 0; v < 8; ++v) {
                float e = __expf(acc[jt][v] - rmax[v]);
                rsum[v] += e;
                Pw[(rbase + v) * QSTR + jt * 16 + col] = (bf16)e;
            }
        }
#pragma unroll
        for (int m = 1; m < 16; m <<= 1) {
#pragma unroll
            for (int v = 0; v < 8; ++v)
                rsum[v] += __shfl_xor(rsum[v], m, 32);
        }
        float rinv[8];
#pragma unroll
        for (int v = 0; v < 8; ++v) rinv[v] = 1.0f / rsum[v];

        // out = P @ V : K=256 in 8 chunks of 32, N=32 channels in 2 tiles
        v8f oc0 = {0.f,0.f,0.f,0.f,0.f,0.f,0.f,0.f};
        v8f oc1 = {0.f,0.f,0.f,0.f,0.f,0.f,0.f,0.f};
        const int prow = lane & 15;
#pragma unroll
        for (int jc = 0; jc < 8; ++jc) {
            AFrag ap;
#pragma unroll
            for (int v = 0; v < 8; ++v) {
                int k0 = jc * 32 + kbA + (v << 1);
                ap.u[v] = *(const unsigned*)&Pw[prow * QSTR + k0];
            }
            AFrag b0, b1;
#pragma unroll
            for (int v = 0; v < 8; ++v) {
                int kk = jc * 32 + kbB + 2 * v;
                b0.u[v] = *(const unsigned*)&Vs[col * QSTR + kk];
                b1.u[v] = *(const unsigned*)&Vs[(col + 16) * QSTR + kk];
            }
            oc0 = __builtin_amdgcn_wmma_f32_16x16x32_bf16(
                      false, ap.v, false, b0.v, (short)0, oc0, false, false);
            oc1 = __builtin_amdgcn_wmma_f32_16x16x32_bf16(
                      false, ap.v, false, b1.v, (short)0, oc1, false, false);
        }

        // normalize by 1/rowsum (same D-register row layout), scatter (b,c,h,w)
#pragma unroll
        for (int v = 0; v < 8; ++v) {
            int qidx = q0 + rbase + v;
            int h = hd * 16 + (qidx >> 4);
            int w = wd * 16 + (qidx & 15);
            size_t base = ((size_t)bi * 192 + head * 32) * 36864
                        + (size_t)h * 192 + w;
            outb[base + (size_t)col * 36864]        = (bf16)(oc0[v] * rinv[v]);
            outb[base + (size_t)(col + 16) * 36864] = (bf16)(oc1[v] * rinv[v]);
        }
    }
}

// ---------------------------------------------------------------------------
// Host launcher
// ---------------------------------------------------------------------------
extern "C" void kernel_launch(void* const* d_in, const int* in_sizes, int n_in,
                              void* d_out, int out_size, void* d_ws, size_t ws_size,
                              hipStream_t stream) {
    const float* x     = (const float*)d_in[0];  // (2,192,192,192)
    const float* w_qkv = (const float*)d_in[1];  // (576,192)
    const float* b_qkv = (const float*)d_in[2];  // (576)
    const float* rpe   = (const float*)d_in[3];  // (6,961)
    const float* w_out = (const float*)d_in[4];  // (192,192)
    const float* b_out = (const float*)d_in[5];  // (192)
    float* out = (float*)d_out;

    const size_t HW = 36864;  // 192*192
    char* ws = (char*)d_ws;
    size_t off = 0;
    auto carve = [&](size_t bytes) {
        void* p = ws + off;
        off += (bytes + 255) & ~(size_t)255;
        return p;
    };
    bf16* x_b    = (bf16*)carve(2ull * 192 * HW * 2);
    bf16* qkv_b  = (bf16*)carve(2ull * 576 * HW * 2);
    bf16* attn_b = (bf16*)carve(2ull * 192 * HW * 2);
    bf16* wqkv_b = (bf16*)carve(576ull * 192 * 2);
    bf16* wout_b = (bf16*)carve(192ull * 192 * 2);

    // 0) precision conversion
    f32_to_bf16_kernel<<<(2 * 192 * 36864) / 256, 256, 0, stream>>>(
        x, x_b, 2 * 192 * 36864);
    f32_to_bf16_kernel<<<(576 * 192 + 255) / 256, 256, 0, stream>>>(
        w_qkv, wqkv_b, 576 * 192);
    f32_to_bf16_kernel<<<(192 * 192 + 255) / 256, 256, 0, stream>>>(
        w_out, wout_b, 192 * 192);

    // 1) QKV projection: (576x192) @ (192x36864) per batch -> bf16
    gemm_wmma_bf16<false><<<dim3(36864 / 128, 576 / 64, 2), 256, 0, stream>>>(
        wqkv_b, x_b, b_qkv, qkv_b, nullptr, 576, 192, 36864);

    // 2) window attention: 2 * 12*12 windows * 6 heads = 1728 blocks
    const size_t attn_lds = (size_t)(3 * 32 * QSTR + 8 * 16 * QSTR) * 2
                          + 961 * 4 + 16;
    attn_wmma_kernel<<<dim3(1728), 256, attn_lds, stream>>>(qkv_b, rpe, attn_b);

    // 3) output projection: (192x192) @ (192x36864) per batch -> fp32 d_out
    gemm_wmma_bf16<true><<<dim3(36864 / 128, 192 / 64, 2), 256, 0, stream>>>(
        wout_b, attn_b, b_out, nullptr, out, 192, 192, 36864);
}
